// MHA_24919400252086
// MI455X (gfx1250) — matrix operand order repaired and checked by reading, not compiled, not run on hardware
//
#include <hip/hip_runtime.h>

typedef __attribute__((ext_vector_type(16))) _Float16 v16h;
typedef __attribute__((ext_vector_type(8)))  _Float16 v8h;
typedef __attribute__((ext_vector_type(4)))  float    v4f;
typedef __attribute__((ext_vector_type(8)))  float    v8f;

static constexpr int kD    = 1024;            // D_MODEL == INNER
static constexpr int kH    = 16;
static constexpr int kDh   = 64;
static constexpr int kT    = 2048;
static constexpr int kB    = 4;
static constexpr int kRows = kB * kT;         // 8192 flattened tokens

__device__ __forceinline__ float redmax16(float v) {
  v = fmaxf(v, __shfl_xor(v, 1, 32));
  v = fmaxf(v, __shfl_xor(v, 2, 32));
  v = fmaxf(v, __shfl_xor(v, 4, 32));
  v = fmaxf(v, __shfl_xor(v, 8, 32));
  return v;
}
__device__ __forceinline__ float redsum16(float v) {
  v += __shfl_xor(v, 1, 32);
  v += __shfl_xor(v, 2, 32);
  v += __shfl_xor(v, 4, 32);
  v += __shfl_xor(v, 8, 32);
  return v;
}

// ---------------------------------------------------------------------------
// Weight prep: Wt[n][k] = (f16) W[k][n].  LDS-tiled so both global sides are
// coalesced. One 32x32 tile per block (32x8 threads).
// ---------------------------------------------------------------------------
__global__ void transpose_w(const float* __restrict__ W, _Float16* __restrict__ Wt) {
  __shared__ float tile[32][33];
  const int tx = threadIdx.x;           // 0..31
  const int ty = threadIdx.y;           // 0..7
  const int n0 = blockIdx.x * 32;
  const int k0 = blockIdx.y * 32;
#pragma unroll
  for (int r = ty; r < 32; r += 8)
    tile[r][tx] = W[(size_t)(k0 + r) * kD + n0 + tx];
  __syncthreads();
#pragma unroll
  for (int r = ty; r < 32; r += 8)
    Wt[(size_t)(n0 + r) * kD + k0 + tx] = (_Float16)tile[tx][r];
}

// ---------------------------------------------------------------------------
// GEMM: 32x64 output tile per wave = 8 x v_wmma_f32_16x16x32_f16 per k-step.
// A: row-major [8192 x 1024], f32 (x, converted inline) or f16 (ctx).
// Wt: f16, transposed [N][K] -> B fragments are contiguous b128 loads.
// vmode: 0 = f32 row-major store to outF (output projection)
//        1 = f16 head-major [B,H,T,Dh] store (Q,K)
//        2 = f16 head-transposed [B,H,Dh,T] store (V)
// ---------------------------------------------------------------------------
template <typename TA>
__global__ void gemm16_wmma(const TA* __restrict__ A, const _Float16* __restrict__ Wt,
                            _Float16* __restrict__ outH, float* __restrict__ outF,
                            int vmode) {
  const int lane = threadIdx.x;
  const int l15  = lane & 15;
  const int hi   = lane >> 4;          // 0: lanes 0-15, 1: lanes 16-31
  const int m0   = blockIdx.x * 32;    // two 16-row tiles
  const int n0   = blockIdx.y * 64;    // four 16-col tiles

  v8f acc[2][4];
#pragma unroll
  for (int r = 0; r < 2; ++r)
#pragma unroll
    for (int g = 0; g < 4; ++g)
#pragma unroll
      for (int i = 0; i < 8; ++i) acc[r][g][i] = 0.f;

  const TA* Arow[2] = { A + (size_t)(m0 + l15) * kD,
                        A + (size_t)(m0 + 16 + l15) * kD };
  const int asel = hi ? 8 : 0;

  for (int k0 = 0; k0 < kD; k0 += 32) {
    // A fragments (16x32 f16 each)
    v16h a[2];
#pragma unroll
    for (int r = 0; r < 2; ++r) {
      if constexpr (sizeof(TA) == 4) {
        v4f r0 = *(const v4f*)(Arow[r] + k0 + asel);
        v4f r1 = *(const v4f*)(Arow[r] + k0 + asel + 4);
        v4f r2 = *(const v4f*)(Arow[r] + k0 + 16 + asel);
        v4f r3 = *(const v4f*)(Arow[r] + k0 + 16 + asel + 4);
#pragma unroll
        for (int i = 0; i < 4; ++i) {
          a[r][i]      = (_Float16)r0[i];
          a[r][4 + i]  = (_Float16)r1[i];
          a[r][8 + i]  = (_Float16)r2[i];
          a[r][12 + i] = (_Float16)r3[i];
        }
      } else {
        v8h r0 = *(const v8h*)(Arow[r] + k0 + asel);
        v8h r1 = *(const v8h*)(Arow[r] + k0 + 16 + asel);
#pragma unroll
        for (int i = 0; i < 8; ++i) { a[r][i] = r0[i]; a[r][8 + i] = r1[i]; }
      }
    }

    const int kb = k0 + (hi ? 8 : 0);    // B-layout: table order, matches A
#pragma unroll
    for (int g = 0; g < 4; ++g) {
      // N = n0 + g*16 + (lane&15); K = kb..kb+7 and kb+16..kb+23 in Wt
      const _Float16* Wp = Wt + (size_t)(n0 + g * 16 + l15) * kD + kb;
      v16h b;
      v8h b0 = *(const v8h*)(Wp);
      v8h b1 = *(const v8h*)(Wp + 16);
#pragma unroll
      for (int i = 0; i < 8; ++i) { b[i] = b0[i]; b[8 + i] = b1[i]; }
      acc[0][g] = __builtin_amdgcn_wmma_f32_16x16x32_f16(
          false, a[0], false, b, (short)0, acc[0][g], false, false);
      acc[1][g] = __builtin_amdgcn_wmma_f32_16x16x32_f16(
          false, a[1], false, b, (short)0, acc[1][g], false, false);
    }
  }

  // C-layout store: VGPR i -> row i (lanes 0-15) / row i+8 (lanes 16-31)
#pragma unroll
  for (int r = 0; r < 2; ++r) {
#pragma unroll
    for (int g = 0; g < 4; ++g) {
#pragma unroll
      for (int i = 0; i < 8; ++i) {
        const int m = m0 + r * 16 + i + (hi ? 8 : 0);
        const int n = n0 + g * 16 + l15;
        const float v = acc[r][g][i];
        if (vmode == 0) {
          outF[(size_t)m * kD + n] = v;
        } else {
          const int bb = m >> 11, t = m & (kT - 1);
          const int hh = n >> 6,  dd = n & (kDh - 1);
          if (vmode == 1)        // [B][H][T][Dh]
            outH[(((size_t)(bb * kH + hh)) * kT + t) * kDh + dd] = (_Float16)v;
          else                   // [B][H][Dh][T]
            outH[(((size_t)(bb * kH + hh)) * kDh + dd) * kT + t] = (_Float16)v;
        }
      }
    }
  }
}

// ---------------------------------------------------------------------------
// Causal flash attention: one wave per (b, h, 16-query tile).
// Q,K f16 [B,H,T,Dh]; V f16 transposed [B,H,Dh,T]; ctx f16 [B*T, 1024].
// ---------------------------------------------------------------------------
__global__ void attn_wmma(const _Float16* __restrict__ Q, const _Float16* __restrict__ K,
                          const _Float16* __restrict__ V, _Float16* __restrict__ C) {
  __shared__ _Float16 pTile[16 * 32];   // P staging: C-layout -> A-layout
  const int lane  = threadIdx.x;
  const int l15   = lane & 15;
  const int hi    = lane >> 4;
  const int qt    = blockIdx.x;         // query tile
  const int bh    = blockIdx.y;         // b*16 + h
  const int bb    = bh >> 4;
  const int hh    = bh & 15;
  const int qbase = qt * 16;
  const size_t headOff = (size_t)bh * kT * kDh;   // same element count both layouts
  const int asel = hi ? 8 : 0;
  const int ksel = hi ? 16 : 0;

  // Q A-fragments: 16x64 as two 16x32 chunks
  v16h aq[2];
  {
    const _Float16* Qrow = Q + headOff + (size_t)(qbase + l15) * kDh;
#pragma unroll
    for (int c = 0; c < 2; ++c) {
      v8h r0 = *(const v8h*)(Qrow + c * 32 + asel);
      v8h r1 = *(const v8h*)(Qrow + c * 32 + 16 + asel);
#pragma unroll
      for (int i = 0; i < 8; ++i) { aq[c][i] = r0[i]; aq[c][8 + i] = r1[i]; }
    }
  }

  v8f o[4];
#pragma unroll
  for (int g = 0; g < 4; ++g)
#pragma unroll
    for (int i = 0; i < 8; ++i) o[g][i] = 0.f;
  float mrow[8], lrow[8];
#pragma unroll
  for (int i = 0; i < 8; ++i) { mrow[i] = -3.0e38f; lrow[i] = 0.f; }

  for (int kv0 = 0; kv0 < qbase + 16; kv0 += 32) {
    // ---- S = Q * K^T for 32 keys (two 16x16 tiles) ----
    v8f s[2];
#pragma unroll
    for (int j = 0; j < 2; ++j) {
      const _Float16* Krow = K + headOff + (size_t)(kv0 + j * 16 + l15) * kDh + asel;
      v16h b0, b1;
      v8h x0 = *(const v8h*)(Krow);
      v8h x1 = *(const v8h*)(Krow + 16);
      v8h y0 = *(const v8h*)(Krow + 32);
      v8h y1 = *(const v8h*)(Krow + 48);
#pragma unroll
      for (int i = 0; i < 8; ++i) {
        b0[i] = x0[i]; b0[8 + i] = x1[i];
        b1[i] = y0[i]; b1[8 + i] = y1[i];
      }
      v8f z;
#pragma unroll
      for (int i = 0; i < 8; ++i) z[i] = 0.f;
      z = __builtin_amdgcn_wmma_f32_16x16x32_f16(false, aq[0], false, b0, (short)0, z, false, false);
      z = __builtin_amdgcn_wmma_f32_16x16x32_f16(false, aq[1], false, b1, (short)0, z, false, false);
      s[j] = z;
    }

    // ---- causal mask + online softmax (per-row reductions in 16-lane halves) ----
    float p0[8], p1[8];
#pragma unroll
    for (int i = 0; i < 8; ++i) {
      const int qg = qbase + i + (hi ? 8 : 0);
      float s0 = (float)s[0][i] * 0.125f;   // 1/sqrt(64)
      float s1 = (float)s[1][i] * 0.125f;
      if (kv0 + l15 > qg)      s0 = -3.0e38f;
      if (kv0 + 16 + l15 > qg) s1 = -3.0e38f;
      const float rm   = redmax16(fmaxf(s0, s1));
      const float mnew = fmaxf(mrow[i], rm);
      const float corr = __expf(mrow[i] - mnew);
      const float e0   = __expf(s0 - mnew);
      const float e1   = __expf(s1 - mnew);
      const float rs   = redsum16(e0 + e1);
      lrow[i] = lrow[i] * corr + rs;
      mrow[i] = mnew;
#pragma unroll
      for (int g = 0; g < 4; ++g) o[g][i] *= corr;
      p0[i] = e0; p1[i] = e1;
    }

    // ---- P: C-layout -> LDS (row major 16x32) -> A-layout fragment ----
    __syncthreads();
#pragma unroll
    for (int i = 0; i < 8; ++i) {
      const int r = i + (hi ? 8 : 0);
      pTile[r * 32 + l15]      = (_Float16)p0[i];
      pTile[r * 32 + 16 + l15] = (_Float16)p1[i];
    }
    __syncthreads();
    v16h ap;
    {
      v8h r0 = *(const v8h*)(&pTile[l15 * 32 + asel]);
      v8h r1 = *(const v8h*)(&pTile[l15 * 32 + 16 + asel]);
#pragma unroll
      for (int i = 0; i < 8; ++i) { ap[i] = r0[i]; ap[8 + i] = r1[i]; }
    }

    // ---- O += P * V  via transposed V: contiguous b128 B-fragment loads ----
#pragma unroll
    for (int g = 0; g < 4; ++g) {
      // N = out dim d = g*16 + (lane&15); K = key index asel..asel+7, 16+asel..16+asel+7
      const _Float16* Vp = V + headOff + (size_t)(g * 16 + l15) * kT + kv0 + asel;
      v16h bv;
      v8h r0 = *(const v8h*)(Vp);
      v8h r1 = *(const v8h*)(Vp + 16);
#pragma unroll
      for (int i = 0; i < 8; ++i) { bv[i] = r0[i]; bv[8 + i] = r1[i]; }
      o[g] = __builtin_amdgcn_wmma_f32_16x16x32_f16(
          false, ap, false, bv, (short)0, o[g], false, false);
    }
  }

  // ---- normalize + store ctx f16 row-major [B*T, 1024] ----
#pragma unroll
  for (int g = 0; g < 4; ++g) {
#pragma unroll
    for (int i = 0; i < 8; ++i) {
      const int t = qbase + i + (hi ? 8 : 0);
      const int d = hh * kDh + g * 16 + l15;
      C[((size_t)bb * kT + t) * kD + d] = (_Float16)(o[g][i] / lrow[i]);
    }
  }
}

// ---------------------------------------------------------------------------
extern "C" void kernel_launch(void* const* d_in, const int* in_sizes, int n_in,
                              void* d_out, int out_size, void* d_ws, size_t ws_size,
                              hipStream_t stream) {
  (void)in_sizes; (void)n_in; (void)out_size; (void)ws_size;
  const float* x  = (const float*)d_in[0];
  const float* Wq = (const float*)d_in[1];
  const float* Wk = (const float*)d_in[2];
  const float* Wv = (const float*)d_in[3];
  const float* Wo = (const float*)d_in[4];

  const size_t actElems = (size_t)kRows * kD;   // 8.39M f16 each
  const size_t wElems   = (size_t)kD * kD;      // 1.05M f16 each
  _Float16* Qh  = (_Float16*)d_ws;              // [B,H,T,Dh]
  _Float16* Kh  = Qh + actElems;                // [B,H,T,Dh]
  _Float16* Vt  = Kh + actElems;                // [B,H,Dh,T]
  _Float16* Ch  = Vt + actElems;                // ctx [B*T, 1024]
  _Float16* Wqt = Ch + actElems;                // f16 W^T [N][K]
  _Float16* Wkt = Wqt + wElems;
  _Float16* Wvt = Wkt + wElems;
  _Float16* Wot = Wvt + wElems;                 // total ~75.5 MB

  const dim3 tGrid(kD / 32, kD / 32), tBlk(32, 8);
  transpose_w<<<tGrid, tBlk, 0, stream>>>(Wq, Wqt);
  transpose_w<<<tGrid, tBlk, 0, stream>>>(Wk, Wkt);
  transpose_w<<<tGrid, tBlk, 0, stream>>>(Wv, Wvt);
  transpose_w<<<tGrid, tBlk, 0, stream>>>(Wo, Wot);

  const dim3 gGemm(kRows / 32, kD / 64);        // (256, 16), 1 wave per block
  gemm16_wmma<float><<<gGemm, 32, 0, stream>>>(x, Wqt, Qh, nullptr, 1);
  gemm16_wmma<float><<<gGemm, 32, 0, stream>>>(x, Wkt, Kh, nullptr, 1);
  gemm16_wmma<float><<<gGemm, 32, 0, stream>>>(x, Wvt, Vt, nullptr, 2);

  attn_wmma<<<dim3(kT / 16, kB * kH), 32, 0, stream>>>(Qh, Kh, Vt, Ch);

  gemm16_wmma<_Float16><<<gGemm, 32, 0, stream>>>(Ch, Wot, nullptr, (float*)d_out, 0);
}
